// LSTMNet_28922309771279
// MI455X (gfx1250) — matrix operand order repaired and checked by reading, not compiled, not run on hardware
//
#include <hip/hip_runtime.h>

// ---------------- problem constants ----------------
#define Tn   512
#define Bn   1024
#define Hn   128
#define INn  4
#define FUTn 16
#define TOUT (Tn + FUTn)
#define BM   32              // batch rows per workgroup
#define NWG  (Bn / BM)       // 32 workgroups

typedef __attribute__((ext_vector_type(16))) __bf16 v16bf;
typedef __attribute__((ext_vector_type(8)))  float  v8f;

struct alignas(16) Q4 { unsigned int a, b, c, d; };

__device__ __forceinline__ float sigf(float x) {
    return 1.0f / (1.0f + __expf(-x));
}
__device__ __forceinline__ float tanh_(float x) {
    // overflow-safe tanh via exp
    float ax = fabsf(x);
    float e  = __expf(-2.0f * ax);
    float r  = (1.0f - e) / (1.0f + e);
    return x < 0.0f ? -r : r;
}

// A fragment: 16(M) x 32(K) bf16, row-major source with row stride Hn.
// Lane layout (CDNA5 ISA 7.12.2): lanes 0-15 -> rows 0-15, K = {0..7, 16..23};
// lanes 16-31 -> rows 0-15, K = {8..15, 24..31}.
__device__ __forceinline__ v16bf loadA(const __bf16* base, int lane) {
    int r  = lane & 15;
    int hi = lane >> 4;
    const __bf16* p = base + r * Hn + hi * 8;
    union { Q4 q[2]; v16bf v; } u;
    u.q[0] = *(const Q4*)(p);
    u.q[1] = *(const Q4*)(p + 16);
    return u.v;
}

// B fragment: 32(K) x 16(N) bf16 where B[k][n] = W[n][k], W row-major [512][Hn].
// Lanes 0-15 -> col n, K = {0..7, 16..23}; lanes 16-31 -> col n, K = {8..15, 24..31}.
__device__ __forceinline__ v16bf loadB(const __bf16* w, int n0, int k0, int lane) {
    int c  = lane & 15;
    int hi = lane >> 4;
    const __bf16* p = w + (size_t)(n0 + c) * Hn + k0 + hi * 8;
    union { Q4 q[2]; v16bf v; } u;
    u.q[0] = *(const Q4*)(p);
    u.q[1] = *(const Q4*)(p + 16);
    return u.v;
}

#define WMMA_BF16(A, B, C) \
    __builtin_amdgcn_wmma_f32_16x16x32_bf16(false, (A), false, (B), (short)0, (C), false, false)

// ---------------- weight conversion prologue ----------------
__global__ void lstm_cvt_weights(const float* __restrict__ whh0,
                                 const float* __restrict__ wih1,
                                 const float* __restrict__ whh1,
                                 __bf16* __restrict__ out) {
    const int N = 512 * Hn;
    int i = blockIdx.x * blockDim.x + threadIdx.x;
    if (i < N)            out[i] = (__bf16)whh0[i];
    else if (i < 2 * N)   out[i] = (__bf16)wih1[i - N];
    else if (i < 3 * N)   out[i] = (__bf16)whh1[i - 2 * N];
}

// ---------------- persistent LSTM kernel ----------------
__global__ __launch_bounds__(256)
void lstm_persistent(const float* __restrict__ x,      // [B, IN, T]
                     const float* __restrict__ wih0,   // [512, 4] f32
                     const float* __restrict__ bih0, const float* __restrict__ bhh0,
                     const float* __restrict__ bih1, const float* __restrict__ bhh1,
                     const float* __restrict__ wl,     // [4, 128] f32
                     const float* __restrict__ bl,     // [4] f32
                     const __bf16* __restrict__ whh0g, // [512,128] bf16
                     const __bf16* __restrict__ wih1g, // [512,128] bf16
                     const __bf16* __restrict__ whh1g, // [512,128] bf16
                     float* __restrict__ out)          // [B, 4, TOUT]
{
    extern __shared__ char smem[];
    __bf16* s_whh0 = (__bf16*)smem;                 // 512*128
    __bf16* s_whh1 = s_whh0 + 512 * Hn;             // 512*128
    __bf16* s_h0   = s_whh1 + 512 * Hn;             // BM*128
    __bf16* s_h1   = s_h0 + BM * Hn;                // BM*128
    float*  s_w0   = (float*)(s_h1 + BM * Hn);      // 512*4
    float*  s_b0   = s_w0 + 512 * 4;                // 512
    float*  s_b1   = s_b0 + 512;                    // 512
    float*  s_wl   = s_b1 + 512;                    // 4*128
    float*  s_bl   = s_wl + 4 * Hn;                 // 4
    float*  s_x    = s_bl + 4;                      // BM*4
    float*  s_xf   = s_x + BM * 4;                  // BM*4

    const int tid    = threadIdx.x;
    const int wv     = tid >> 5;
    const int lane   = tid & 31;
    const int hi     = lane >> 4;
    const int col    = (wv << 4) + (lane & 15);     // hidden column this lane owns
    const int wgBase = blockIdx.x * BM;

    // ---- stage weights / constants into LDS ----
    {
        const Q4* s0 = (const Q4*)whh0g;  Q4* d0 = (Q4*)s_whh0;
        const Q4* s1 = (const Q4*)whh1g;  Q4* d1 = (Q4*)s_whh1;
        const int nq = 512 * Hn / 8;
        for (int i = tid; i < nq; i += 256) { d0[i] = s0[i]; d1[i] = s1[i]; }
    }
    for (int i = tid; i < 512 * 4; i += 256) s_w0[i] = wih0[i];
    for (int i = tid; i < 512; i += 256) {
        s_b0[i] = bih0[i] + bhh0[i];
        s_b1[i] = bih1[i] + bhh1[i];
    }
    for (int i = tid; i < 4 * Hn; i += 256) s_wl[i] = wl[i];
    if (tid < 4) s_bl[tid] = bl[tid];
    for (int i = tid; i < BM * Hn; i += 256) {
        s_h0[i] = (__bf16)0.0f;
        s_h1[i] = (__bf16)0.0f;
    }

    // register-resident cell state: c[mt][e], hidden column `col`
    v8f c0r[2], c1r[2];
#pragma unroll
    for (int mt = 0; mt < 2; ++mt)
#pragma unroll
        for (int e = 0; e < 8; ++e) { c0r[mt][e] = 0.0f; c1r[mt][e] = 0.0f; }

    __syncthreads();

    for (int t = 0; t < TOUT; ++t) {
        const bool futu = (t >= Tn);

        // ---- phase 1: stage x_t (or carried output) ----
        if (!futu) {
            if (tid < BM * 4) {
                int b = tid >> 2, k = tid & 3;
                s_x[tid] = x[(size_t)(wgBase + b) * (INn * Tn) + k * Tn + t];
            }
        } else if (t == Tn) {
            if (tid < BM * 4) s_x[tid] = s_xf[tid];   // seed from out[T-2]
        }
        __syncthreads();

        // =========== layer 0 gates ===========
        v8f acc[2][4];
#pragma unroll
        for (int mt = 0; mt < 2; ++mt)
#pragma unroll
            for (int j = 0; j < 4; ++j) {
                float bb = s_b0[j * Hn + col];
#pragma unroll
                for (int e = 0; e < 8; ++e) acc[mt][j][e] = bb;
            }
        // small input term: x_t[b,0..3] * Wih0[n,0..3]
#pragma unroll
        for (int mt = 0; mt < 2; ++mt) {
            float xr[8][4];
#pragma unroll
            for (int e = 0; e < 8; ++e) {
                int br = mt * 16 + hi * 8 + e;
#pragma unroll
                for (int k = 0; k < 4; ++k) xr[e][k] = s_x[br * 4 + k];
            }
#pragma unroll
            for (int j = 0; j < 4; ++j) {
                int n = j * Hn + col;
                float w0 = s_w0[n * 4 + 0], w1 = s_w0[n * 4 + 1];
                float w2 = s_w0[n * 4 + 2], w3 = s_w0[n * 4 + 3];
#pragma unroll
                for (int e = 0; e < 8; ++e)
                    acc[mt][j][e] += xr[e][0] * w0 + xr[e][1] * w1
                                   + xr[e][2] * w2 + xr[e][3] * w3;
            }
        }
        // h0 @ Whh0^T  (LDS x LDS)
#pragma unroll
        for (int kc = 0; kc < 4; ++kc) {
            v16bf a0 = loadA(s_h0 + kc * 32, lane);
            v16bf a1 = loadA(s_h0 + 16 * Hn + kc * 32, lane);
#pragma unroll
            for (int j = 0; j < 4; ++j) {
                int nt = j * 8 + wv;
                v16bf b = loadB(s_whh0, nt * 16, kc * 32, lane);
                acc[0][j] = WMMA_BF16(a0, b, acc[0][j]);
                acc[1][j] = WMMA_BF16(a1, b, acc[1][j]);
            }
        }
        __syncthreads();   // all h0 reads complete

        // cell update layer 0 (lane-local, c in registers)
#pragma unroll
        for (int mt = 0; mt < 2; ++mt)
#pragma unroll
            for (int e = 0; e < 8; ++e) {
                float ig = sigf(acc[mt][0][e]);
                float fg = sigf(acc[mt][1][e]);
                float gg = tanh_(acc[mt][2][e]);
                float og = sigf(acc[mt][3][e]);
                float c  = fg * c0r[mt][e] + ig * gg;
                c0r[mt][e] = c;
                float h  = og * tanh_(c);
                int row = mt * 16 + hi * 8 + e;
                s_h0[row * Hn + col] = (__bf16)h;
            }
        __syncthreads();   // new h0 visible

        // =========== layer 1 gates ===========
#pragma unroll
        for (int mt = 0; mt < 2; ++mt)
#pragma unroll
            for (int j = 0; j < 4; ++j) {
                float bb = s_b1[j * Hn + col];
#pragma unroll
                for (int e = 0; e < 8; ++e) acc[mt][j][e] = bb;
            }
        // h0 @ Wih1^T  (LDS x global bf16, L2-resident)
#pragma unroll
        for (int kc = 0; kc < 4; ++kc) {
            v16bf a0 = loadA(s_h0 + kc * 32, lane);
            v16bf a1 = loadA(s_h0 + 16 * Hn + kc * 32, lane);
#pragma unroll
            for (int j = 0; j < 4; ++j) {
                int nt = j * 8 + wv;
                v16bf b = loadB(wih1g, nt * 16, kc * 32, lane);
                acc[0][j] = WMMA_BF16(a0, b, acc[0][j]);
                acc[1][j] = WMMA_BF16(a1, b, acc[1][j]);
            }
        }
        // h1 @ Whh1^T  (LDS x LDS)
#pragma unroll
        for (int kc = 0; kc < 4; ++kc) {
            v16bf a0 = loadA(s_h1 + kc * 32, lane);
            v16bf a1 = loadA(s_h1 + 16 * Hn + kc * 32, lane);
#pragma unroll
            for (int j = 0; j < 4; ++j) {
                int nt = j * 8 + wv;
                v16bf b = loadB(s_whh1, nt * 16, kc * 32, lane);
                acc[0][j] = WMMA_BF16(a0, b, acc[0][j]);
                acc[1][j] = WMMA_BF16(a1, b, acc[1][j]);
            }
        }
        __syncthreads();   // all h1 reads complete

        // cell update layer 1
#pragma unroll
        for (int mt = 0; mt < 2; ++mt)
#pragma unroll
            for (int e = 0; e < 8; ++e) {
                float ig = sigf(acc[mt][0][e]);
                float fg = sigf(acc[mt][1][e]);
                float gg = tanh_(acc[mt][2][e]);
                float og = sigf(acc[mt][3][e]);
                float c  = fg * c1r[mt][e] + ig * gg;
                c1r[mt][e] = c;
                float h  = og * tanh_(c);
                int row = mt * 16 + hi * 8 + e;
                s_h1[row * Hn + col] = (__bf16)h;
            }
        __syncthreads();   // new h1 visible

        // ---- output projection H -> 4 ----
        if (tid < BM * 4) {
            int b = tid >> 2, f = tid & 3;
            float s = s_bl[f];
            const __bf16* hr = s_h1 + b * Hn;
            const float*  wr = s_wl + f * Hn;
#pragma unroll 8
            for (int j2 = 0; j2 < Hn; ++j2) s += (float)hr[j2] * wr[j2];
            out[(size_t)(wgBase + b) * (INn * TOUT) + f * TOUT + t] = s;
            if (t == Tn - 2) s_xf[tid] = s;   // seed for autoregression
            if (futu)        s_x[tid]  = s;   // feed next future step
        }
        __syncthreads();
    }
}

extern "C" void kernel_launch(void* const* d_in, const int* in_sizes, int n_in,
                              void* d_out, int out_size, void* d_ws, size_t ws_size,
                              hipStream_t stream) {
    const float* x    = (const float*)d_in[0];
    const float* Wih0 = (const float*)d_in[1];
    const float* Whh0 = (const float*)d_in[2];
    const float* bih0 = (const float*)d_in[3];
    const float* bhh0 = (const float*)d_in[4];
    const float* Wih1 = (const float*)d_in[5];
    const float* Whh1 = (const float*)d_in[6];
    const float* bih1 = (const float*)d_in[7];
    const float* bhh1 = (const float*)d_in[8];
    const float* Wl   = (const float*)d_in[9];
    const float* bl   = (const float*)d_in[10];

    __bf16* wsb   = (__bf16*)d_ws;
    __bf16* whh0b = wsb;
    __bf16* wih1b = wsb + 512 * Hn;
    __bf16* whh1b = wsb + 2 * 512 * Hn;

    // 3 * 512*128 elements -> 768 blocks of 256
    lstm_cvt_weights<<<768, 256, 0, stream>>>(Whh0, Wih1, Whh1, wsb);

    constexpr size_t SMEM_BYTES =
        (size_t)(2 * 512 * Hn + 2 * BM * Hn) * sizeof(__bf16) +
        (size_t)(512 * 4 + 512 + 512 + 4 * Hn + 4 + BM * 4 + BM * 4) * sizeof(float);

    lstm_persistent<<<NWG, 256, SMEM_BYTES, stream>>>(
        x, Wih0, bih0, bhh0, bih1, bhh1, Wl, bl,
        whh0b, wih1b, whh1b, (float*)d_out);
}
